// SwinTransformerBlock_58317065945840
// MI455X (gfx1250) — matrix operand order repaired and checked by reading, not compiled, not run on hardware
//
#include <hip/hip_runtime.h>
#include <hip/hip_bf16.h>
#include <cstdint>

typedef __bf16 bf16;
typedef __bf16 v16bf __attribute__((ext_vector_type(16)));
typedef __bf16 v8bf  __attribute__((ext_vector_type(8)));
typedef float  v8f   __attribute__((ext_vector_type(8)));

#define NTOK 49
#define NPAD 64

// ---------- WMMA fragment loaders (bf16, 16x16x32) ----------
// A operand: row r = lane%16, half h = lane/16.
//   elems 0..7  = K = kbase + h*8 + 0..7 ; elems 8..15 = K = kbase + 16 + h*8 + 0..7
__device__ __forceinline__ v16bf frag_a(const bf16* base, int ld, int kbase, int lane) {
    int r = lane & 15, h = lane >> 4;
    const bf16* p = base + r * ld + kbase + h * 8;
    v8bf lo = *(const v8bf*)p;
    v8bf hi = *(const v8bf*)(p + 16);
    v16bf f;
#pragma unroll
    for (int i = 0; i < 8; ++i) { f[i] = lo[i]; f[i + 8] = hi[i]; }
    return f;
}
// B operand: col n = lane%16, half h = lane/16; elems 0..7 = K = kbase + h*8 + 0..7 ; elems 8..15 = K = kbase + 16 + h*8 + 0..7
// (source stored transposed: [n][k], stride ld)
__device__ __forceinline__ v16bf frag_b(const bf16* base, int ld, int kbase, int lane) {
    int n = lane & 15, h = lane >> 4;
    const bf16* p = base + n * ld + kbase + h * 8;
    v8bf lo = *(const v8bf*)p;
    v8bf hi = *(const v8bf*)(p + 16);
    v16bf f;
#pragma unroll
    for (int i = 0; i < 8; ++i) { f[i] = lo[i]; f[i + 8] = hi[i]; }
    return f;
}
__device__ __forceinline__ v8f wmma_bf(v16bf a, v16bf b, v8f c) {
    return __builtin_amdgcn_wmma_f32_16x16x32_bf16(false, a, false, b, (short)0, c, false, false);
}

// ---------- async global->LDS staging (CDNA5 ASYNCcnt path) ----------
__device__ __forceinline__ void stage_async(bf16* lds, const bf16* gsrc,
                                            int nchunks, int chunk_per_row,
                                            int src_stride_bytes, int tid, int nthr) {
    for (int c = tid; c < nchunks; c += nthr) {
        int row = c / chunk_per_row;
        int col = c - row * chunk_per_row;
        unsigned laddr = (unsigned)(uintptr_t)lds + (unsigned)c * 16u;
        unsigned long long ga = (unsigned long long)(uintptr_t)gsrc +
                                (unsigned long long)row * (unsigned)src_stride_bytes +
                                (unsigned long long)col * 16u;
        asm volatile("global_load_async_to_lds_b128 %0, %1, off"
                     :: "v"(laddr), "v"(ga) : "memory");
    }
    asm volatile("s_wait_asynccnt 0x0" ::: "memory");
}

// ---------- K0: convert weight [K][N] f32 -> transposed bf16 [N][K] ----------
__global__ void k_wcvt(const float* __restrict__ src, bf16* __restrict__ dst, int K, int N) {
    int i = blockIdx.x * 256 + threadIdx.x;
    if (i < K * N) { int n = i % N, k = i / N; dst[n * K + k] = (bf16)src[i]; }
}

// ---------- K1: LN1 + cyclic shift + window partition -> Xw bf16 [2048][64][128] ----------
__global__ void k_ln_part(const float* __restrict__ x, const float* __restrict__ g,
                          const float* __restrict__ bt, bf16* __restrict__ Xw) {
    int w = blockIdx.x;
    int b = w >> 6, wi = w & 63, wy = wi >> 3, wx = wi & 7;
    int wave = __builtin_amdgcn_readfirstlane(threadIdx.x >> 5);
    int lane = threadIdx.x & 31;
    for (int j = wave; j < NPAD; j += 8) {
        bf16* orow = Xw + ((long)w * NPAD + j) * 128;
        if (j >= NTOK) {
#pragma unroll
            for (int q = 0; q < 4; ++q) orow[lane * 4 + q] = (bf16)0.0f;
            continue;
        }
        int ty = j / 7, tx = j % 7;
        int hs = (wy * 7 + ty + 3) % 56;
        int ws_ = (wx * 7 + tx + 3) % 56;
        const float* px = x + (((long)b * 56 + hs) * 56 + ws_) * 128;
        float v0[4]; float s = 0.f, ss = 0.f;
#pragma unroll
        for (int q = 0; q < 4; ++q) { float t = px[lane * 4 + q]; v0[q] = t; s += t; ss += t * t; }
        for (int m = 16; m >= 1; m >>= 1) { s += __shfl_xor(s, m, 32); ss += __shfl_xor(ss, m, 32); }
        float mu = s * (1.f / 128.f);
        float rs = rsqrtf(ss * (1.f / 128.f) - mu * mu + 1e-5f);
#pragma unroll
        for (int q = 0; q < 4; ++q) {
            int c = lane * 4 + q;
            orow[c] = (bf16)((v0[q] - mu) * rs * g[c] + bt[c]);
        }
    }
}

// ---------- K2: QKV GEMM, grid(2048,2), LDS-staged weight half (48KB) ----------
__global__ void k_qkv(const bf16* __restrict__ Xw, const bf16* __restrict__ WqkvT,
                      const float* __restrict__ bqkv, bf16* __restrict__ Q,
                      bf16* __restrict__ Kt, bf16* __restrict__ VT) {
    __shared__ bf16 Wl[192 * 128];            // 12 n-tiles x 128 K  (48KB)
    int w = blockIdx.x;
    int ny = blockIdx.y;
    int tid = threadIdx.x;
    int wave = __builtin_amdgcn_readfirstlane(tid >> 5);   // wave-uniform (SGPR)
    int lane = tid & 31;
    int r = lane & 15, h = lane >> 4;

    stage_async(Wl, WqkvT + (long)ny * 192 * 128, 3072, 3072, 0, tid, 256);
    __syncthreads();

    int mt = wave >> 1;
    int ntl0 = (wave & 1) * 6;
    const bf16* A = Xw + ((long)w * NPAD + mt * 16) * 128;
    v16bf af[4];
#pragma unroll
    for (int kk = 0; kk < 4; ++kk) af[kk] = frag_a(A, 128, kk * 32, lane);

    for (int ntl = ntl0; ntl < ntl0 + 6; ++ntl) {
        v8f acc = {};
        const bf16* Bp = Wl + ntl * 16 * 128;
#pragma unroll
        for (int kk = 0; kk < 4; ++kk) acc = wmma_bf(af[kk], frag_b(Bp, 128, kk * 32, lane), acc);
        int nt = ny * 12 + ntl;               // scalar
        int cbase = nt * 16;                  // scalar
        int sel = cbase >> 7;                 // scalar -> scalar branch below
        int head = (cbase >> 5) & 3;
        int hd = (cbase & 31) + r;
        float bb = bqkv[cbase + r];
        if (sel == 0) {
#pragma unroll
            for (int i = 0; i < 8; ++i) {
                int row = mt * 16 + h * 8 + i;
                float v = (acc[i] + bb) * 0.17677669529663687f;   // 1/sqrt(32)
                Q[(((long)w * 4 + head) * NPAD + row) * 32 + hd] = (bf16)v;
            }
        } else if (sel == 1) {
#pragma unroll
            for (int i = 0; i < 8; ++i) {
                int row = mt * 16 + h * 8 + i;
                Kt[(((long)w * 4 + head) * NPAD + row) * 32 + hd] = (bf16)(acc[i] + bb);
            }
        } else {
#pragma unroll
            for (int i = 0; i < 8; ++i) {
                int row = mt * 16 + h * 8 + i;
                VT[(((long)w * 4 + head) * 32 + hd) * NPAD + row] = (bf16)(acc[i] + bb);
            }
        }
    }
}

// ---------- K3: attention per (window,head), K/V tiles async-staged in LDS ----------
__global__ void k_attn(const bf16* __restrict__ Q, const bf16* __restrict__ Kt,
                       const bf16* __restrict__ VT, const float* __restrict__ rel_table,
                       bf16* __restrict__ Ow) {
    __shared__ float relh[169];
    __shared__ bf16 Kl[NPAD * 32];            // 4KB
    __shared__ bf16 Vl[32 * NPAD];            // 4KB
    __shared__ bf16 Pl[4 * 16 * NPAD];        // 8KB
    int bh = blockIdx.x;                      // w*4 + head
    int w = bh >> 2, head = bh & 3;
    int wi = w & 63, wy = wi >> 3, wx = wi & 7;
    int tid = threadIdx.x;
    int wave = __builtin_amdgcn_readfirstlane(tid >> 5);
    int lane = tid & 31;
    int r = lane & 15, h = lane >> 4;

    stage_async(Kl, Kt + (long)bh * NPAD * 32, 256, 256, 0, tid, 128);
    stage_async(Vl, VT + (long)bh * 32 * NPAD, 256, 256, 0, tid, 128);
    for (int t = tid; t < 169; t += 128) relh[t] = rel_table[t * 4 + head];
    __syncthreads();

    int mt = wave;
    const bf16* Qb = Q + ((long)bh * NPAD + mt * 16) * 32;
    v16bf aq = frag_a(Qb, 32, 0, lane);
    v8f s[4];
#pragma unroll
    for (int nt = 0; nt < 4; ++nt) {
        v8f z = {};
        s[nt] = wmma_bf(aq, frag_b(Kl + nt * 16 * 32, 32, 0, lane), z);
    }
#pragma unroll
    for (int nt = 0; nt < 4; ++nt) {
        int j = nt * 16 + r;
        int jy = j / 7, jx = j % 7;
        int sjh = wy * 7 + jy, sjw = wx * 7 + jx;
        int rjh = sjh < 49 ? 0 : (sjh < 53 ? 1 : 2);
        int rjw = sjw < 49 ? 0 : (sjw < 53 ? 1 : 2);
#pragma unroll
        for (int i = 0; i < 8; ++i) {
            int it = mt * 16 + h * 8 + i;
            if (j >= NTOK) { s[nt][i] = -1e30f; continue; }
            if (it < NTOK) {
                int iy = it / 7, ix = it % 7;
                int sih = wy * 7 + iy, siw = wx * 7 + ix;
                int rih = sih < 49 ? 0 : (sih < 53 ? 1 : 2);
                int riw = siw < 49 ? 0 : (siw < 53 ? 1 : 2);
                float m = (rih == rjh && riw == rjw) ? 0.f : -100.f;
                s[nt][i] += relh[(iy - jy + 6) * 13 + (ix - jx + 6)] + m;
            }
        }
    }
#pragma unroll
    for (int i = 0; i < 8; ++i) {
        float mx = fmaxf(fmaxf(s[0][i], s[1][i]), fmaxf(s[2][i], s[3][i]));
        for (int m = 8; m >= 1; m >>= 1) mx = fmaxf(mx, __shfl_xor(mx, m, 32));
        float sum = 0.f;
#pragma unroll
        for (int nt = 0; nt < 4; ++nt) { float e = __expf(s[nt][i] - mx); s[nt][i] = e; sum += e; }
        for (int m = 8; m >= 1; m >>= 1) sum += __shfl_xor(sum, m, 32);
        float inv = 1.f / sum;
#pragma unroll
        for (int nt = 0; nt < 4; ++nt) s[nt][i] *= inv;
    }
    bf16* Pw = Pl + wave * 16 * NPAD;
#pragma unroll
    for (int nt = 0; nt < 4; ++nt)
#pragma unroll
        for (int i = 0; i < 8; ++i)
            Pw[(h * 8 + i) * NPAD + nt * 16 + r] = (bf16)s[nt][i];
    __syncthreads();

    v16bf a0 = frag_a(Pw, NPAD, 0, lane);
    v16bf a1 = frag_a(Pw, NPAD, 32, lane);
#pragma unroll
    for (int nt = 0; nt < 2; ++nt) {
        v8f o = {};
        o = wmma_bf(a0, frag_b(Vl + nt * 16 * NPAD, NPAD, 0, lane), o);
        o = wmma_bf(a1, frag_b(Vl + nt * 16 * NPAD, NPAD, 32, lane), o);
#pragma unroll
        for (int i = 0; i < 8; ++i) {
            int row = mt * 16 + h * 8 + i;
            Ow[((long)w * NPAD + row) * 128 + head * 32 + nt * 16 + r] = (bf16)o[i];
        }
    }
}

// ---------- K4: proj GEMM + window reverse + unshift + residual -> x1 f32 ----------
__global__ void k_proj(const bf16* __restrict__ Ow, const bf16* __restrict__ WpT,
                       const float* __restrict__ bp, const float* __restrict__ x,
                       float* __restrict__ x1) {
    __shared__ bf16 Wl[128 * 128];            // 32KB
    int w = blockIdx.x;
    int b = w >> 6, wi = w & 63, wy = wi >> 3, wx = wi & 7;
    int tid = threadIdx.x;
    int wave = __builtin_amdgcn_readfirstlane(tid >> 5);
    int lane = tid & 31;
    int r = lane & 15, h = lane >> 4;

    stage_async(Wl, WpT, 2048, 2048, 0, tid, 256);
    __syncthreads();

    int mt = wave >> 1;
    int n0 = (wave & 1) * 4;
    const bf16* A = Ow + ((long)w * NPAD + mt * 16) * 128;
    v16bf af[4];
#pragma unroll
    for (int kk = 0; kk < 4; ++kk) af[kk] = frag_a(A, 128, kk * 32, lane);
    for (int nt = n0; nt < n0 + 4; ++nt) {
        v8f acc = {};
#pragma unroll
        for (int kk = 0; kk < 4; ++kk)
            acc = wmma_bf(af[kk], frag_b(Wl + nt * 16 * 128, 128, kk * 32, lane), acc);
        int ch = nt * 16 + r;
        float bb = bp[ch];
#pragma unroll
        for (int i = 0; i < 8; ++i) {
            int tok = mt * 16 + h * 8 + i;
            if (tok < NTOK) {
                int ty = tok / 7, tx = tok % 7;
                int hs = (wy * 7 + ty + 3) % 56;
                int wsx = (wx * 7 + tx + 3) % 56;
                long idx = (((long)b * 56 + hs) * 56 + wsx) * 128 + ch;
                x1[idx] = x[idx] + acc[i] + bb;
            }
        }
    }
}

// ---------- K5: LN2 -> H2 bf16 [100352][128] ----------
__global__ void k_ln2(const float* __restrict__ x1, const float* __restrict__ g,
                      const float* __restrict__ bt, bf16* __restrict__ H2) {
    long row = (long)blockIdx.x * 8 + __builtin_amdgcn_readfirstlane(threadIdx.x >> 5);
    int lane = threadIdx.x & 31;
    const float* px = x1 + row * 128;
    float v0[4]; float s = 0.f, ss = 0.f;
#pragma unroll
    for (int q = 0; q < 4; ++q) { float t = px[lane * 4 + q]; v0[q] = t; s += t; ss += t * t; }
    for (int m = 16; m >= 1; m >>= 1) { s += __shfl_xor(s, m, 32); ss += __shfl_xor(ss, m, 32); }
    float mu = s * (1.f / 128.f);
    float rs = rsqrtf(ss * (1.f / 128.f) - mu * mu + 1e-5f);
#pragma unroll
    for (int q = 0; q < 4; ++q) {
        int c = lane * 4 + q;
        H2[row * 128 + c] = (bf16)((v0[q] - mu) * rs * g[c] + bt[c]);
    }
}

// ---------- K6: fc1 GEMM + GELU, grid(1568,2), LDS-staged weight half (64KB) ----------
__global__ void k_fc1(const bf16* __restrict__ H2, const bf16* __restrict__ W1T,
                      const float* __restrict__ b1, bf16* __restrict__ Hm) {
    __shared__ bf16 Wl[256 * 128];            // 16 n-tiles x 128 K (64KB)
    int ny = blockIdx.y;
    int tid = threadIdx.x;
    int wave = __builtin_amdgcn_readfirstlane(tid >> 5);
    int lane = tid & 31;
    int r = lane & 15, h = lane >> 4;

    stage_async(Wl, W1T + (long)ny * 256 * 128, 4096, 4096, 0, tid, 256);
    __syncthreads();

    long mt = (long)blockIdx.x * 4 + (wave >> 1);
    const bf16* A = H2 + mt * 16 * 128;
    v16bf af[4];
#pragma unroll
    for (int kk = 0; kk < 4; ++kk) af[kk] = frag_a(A, 128, kk * 32, lane);
    int ntl0 = (wave & 1) * 8;
    for (int ntl = ntl0; ntl < ntl0 + 8; ++ntl) {
        v8f acc = {};
#pragma unroll
        for (int kk = 0; kk < 4; ++kk)
            acc = wmma_bf(af[kk], frag_b(Wl + ntl * 16 * 128, 128, kk * 32, lane), acc);
        int ch = ny * 256 + ntl * 16 + r;
        float bb = b1[ch];
#pragma unroll
        for (int i = 0; i < 8; ++i) {
            long row = mt * 16 + h * 8 + i;
            float v = acc[i] + bb;
            v = 0.5f * v * (1.f + erff(v * 0.70710678118654752f));
            Hm[row * 512 + ch] = (bf16)v;
        }
    }
}

// ---------- K7: fc2 GEMM + bias + residual, 2 K-stages of 64KB LDS ----------
__global__ void k_fc2(const bf16* __restrict__ Hm, const bf16* __restrict__ W2T,
                      const float* __restrict__ b2, const float* __restrict__ x1,
                      float* __restrict__ out) {
    __shared__ bf16 Wl[128 * 256];            // all 128 n-rows x 256-K slice (64KB)
    int tid = threadIdx.x;
    int wave = __builtin_amdgcn_readfirstlane(tid >> 5);
    int lane = tid & 31;
    int r = lane & 15, h = lane >> 4;
    long mt = (long)blockIdx.x * 4 + (wave >> 1);
    int n0 = (wave & 1) * 4;

    v8f acc[4] = {};
#pragma unroll 1
    for (int s = 0; s < 2; ++s) {
        if (s) __syncthreads();
        stage_async(Wl, W2T + s * 256, 4096, 32, 512 * 2, tid, 256);
        __syncthreads();
        const bf16* A = Hm + mt * 16 * 512 + s * 256;
        v16bf af[8];
#pragma unroll
        for (int kk = 0; kk < 8; ++kk) af[kk] = frag_a(A, 512, kk * 32, lane);
#pragma unroll
        for (int t = 0; t < 4; ++t) {
            const bf16* Bp = Wl + (n0 + t) * 16 * 256;
#pragma unroll
            for (int kk = 0; kk < 8; ++kk)
                acc[t] = wmma_bf(af[kk], frag_b(Bp, 256, kk * 32, lane), acc[t]);
        }
    }
#pragma unroll
    for (int t = 0; t < 4; ++t) {
        int ch = (n0 + t) * 16 + r;
        float bb = b2[ch];
#pragma unroll
        for (int i = 0; i < 8; ++i) {
            long row = mt * 16 + h * 8 + i;
            long idx = row * 128 + ch;
            out[idx] = acc[t][i] + bb + x1[idx];
        }
    }
}

extern "C" void kernel_launch(void* const* d_in, const int* in_sizes, int n_in,
                              void* d_out, int out_size, void* d_ws, size_t ws_size,
                              hipStream_t stream) {
    const float* x      = (const float*)d_in[0];
    const float* g1     = (const float*)d_in[1];
    const float* b1     = (const float*)d_in[2];
    const float* w_qkv  = (const float*)d_in[3];
    const float* b_qkv  = (const float*)d_in[4];
    const float* w_proj = (const float*)d_in[5];
    const float* b_proj = (const float*)d_in[6];
    const float* rel_t  = (const float*)d_in[7];
    const float* g2     = (const float*)d_in[8];
    const float* b2     = (const float*)d_in[9];
    const float* w_fc1  = (const float*)d_in[10];
    const float* b_fc1  = (const float*)d_in[11];
    const float* w_fc2  = (const float*)d_in[12];
    const float* b_fc2  = (const float*)d_in[13];
    float* out = (float*)d_out;

    char* p = (char*)d_ws;
    bf16* wqkvT = (bf16*)p; p += (size_t)384 * 128 * 2;
    bf16* wprojT = (bf16*)p; p += (size_t)128 * 128 * 2;
    bf16* wfc1T = (bf16*)p; p += (size_t)512 * 128 * 2;
    bf16* wfc2T = (bf16*)p; p += (size_t)128 * 512 * 2;
    bf16* Xw = (bf16*)p; p += (size_t)2048 * 64 * 128 * 2;
    bf16* Qm = (bf16*)p; p += (size_t)2048 * 4 * 64 * 32 * 2;
    bf16* Km = (bf16*)p; p += (size_t)2048 * 4 * 64 * 32 * 2;
    bf16* VT = (bf16*)p; p += (size_t)2048 * 4 * 32 * 64 * 2;
    bf16* Ow = (bf16*)p; p += (size_t)2048 * 64 * 128 * 2;
    float* x1 = (float*)p; p += (size_t)32 * 56 * 56 * 128 * 4;
    bf16* H2 = (bf16*)p; p += (size_t)100352 * 128 * 2;
    bf16* Hm = (bf16*)p; p += (size_t)100352 * 512 * 2;

    k_wcvt<<<(128 * 384 + 255) / 256, 256, 0, stream>>>(w_qkv, wqkvT, 128, 384);
    k_wcvt<<<(128 * 128 + 255) / 256, 256, 0, stream>>>(w_proj, wprojT, 128, 128);
    k_wcvt<<<(128 * 512 + 255) / 256, 256, 0, stream>>>(w_fc1, wfc1T, 128, 512);
    k_wcvt<<<(512 * 128 + 255) / 256, 256, 0, stream>>>(w_fc2, wfc2T, 512, 128);

    k_ln_part<<<2048, 256, 0, stream>>>(x, g1, b1, Xw);
    k_qkv<<<dim3(2048, 2), 256, 0, stream>>>(Xw, wqkvT, b_qkv, Qm, Km, VT);
    k_attn<<<8192, 128, 0, stream>>>(Qm, Km, VT, rel_t, Ow);
    k_proj<<<2048, 256, 0, stream>>>(Ow, wprojT, b_proj, x, x1);
    k_ln2<<<12544, 256, 0, stream>>>(x1, g2, b2, H2);
    k_fc1<<<dim3(1568, 2), 256, 0, stream>>>(H2, wfc1T, b_fc1, Hm);
    k_fc2<<<1568, 256, 0, stream>>>(Hm, wfc2T, b_fc2, x1, out);
}
